// IBQ_46110768890165
// MI455X (gfx1250) — compile-verified, matrix operand order checked by reading, not run on hardware
//
#include <hip/hip_runtime.h>
#include <hip/hip_bf16.h>

// Problem sizes (fixed by reference)
#define BB 4
#define LL 1024
#define MM (BB*LL)      // 4096 rows
#define DIMK 768
#define ZD 256
#define NN 16384
#define NT (NN/16)      // 1024 n-tiles
#define TEMP_INV 100.0f
#define EPSF 1e-5f

typedef __attribute__((ext_vector_type(16))) __bf16 v16bf;
typedef __attribute__((ext_vector_type(8)))  __bf16 v8bf;
typedef __attribute__((ext_vector_type(8)))  float  v8f;
typedef __attribute__((ext_vector_type(4)))  unsigned int v4u;
typedef __attribute__((ext_vector_type(8)))  int v8i;
typedef __attribute__((ext_vector_type(4)))  int v4i;

// ---- WMMA bf16 fragment loaders (ISA 7.12.2 16-bit A/B layout) ----
// lane l: row/col = l&15, h = l>>4 ; 16 bf16 = two contiguous 16B chunks:
//   K in [k0 + h*8, +8) and [k0 + 16 + h*8, +8)
__device__ inline v16bf load_frag_bf16(const __bf16* rowptr, int k0, int h) {
  v8bf lo = *(const v8bf*)(rowptr + k0 + h * 8);
  v8bf hi = *(const v8bf*)(rowptr + k0 + 16 + h * 8);
  return __builtin_shufflevector(lo, hi, 0,1,2,3,4,5,6,7,8,9,10,11,12,13,14,15);
}

__device__ inline v16bf load_frag_f32(const float* __restrict__ rowptr, int k0, int h) {
  v16bf r;
#pragma unroll
  for (int i = 0; i < 8; i++) r[i]     = (__bf16)rowptr[k0 + h * 8 + i];
#pragma unroll
  for (int i = 0; i < 8; i++) r[8 + i] = (__bf16)rowptr[k0 + 16 + h * 8 + i];
  return r;
}

__device__ inline v8f wmma_bf16(v16bf a, v16bf b, v8f c) {
  return __builtin_amdgcn_wmma_f32_16x16x32_bf16(false, a, false, b, (short)0, c, false, false);
}

// ---- TDM: async copy of one contiguous 8KB codebook tile (16 rows x 256 bf16)
// into LDS, described as a 1D tensor of 1024 x 8-byte elements.
// D# layout per CDNA5 ISA ch.8 (group0 128b, group1 256b).
// Toolchain here uses the 6-arg builtin: (g0, g1, g2, g3, g4, cpol).
__device__ inline void tdm_load_tile(const __bf16* gsrc, unsigned int lds_byte_addr) {
  unsigned long long ga = (unsigned long long)(uintptr_t)gsrc;
  v4u g0;
  g0[0] = 1u;                                   // count=1 (valid), user mode
  g0[1] = lds_byte_addr;                        // lds_addr
  g0[2] = (unsigned int)ga;                     // global_addr[31:0]
  g0[3] = (unsigned int)(ga >> 32) | (2u << 30);// global_addr[56:32] | type=2
  v8i g1;
  g1[0] = (int)(3u << 16);                      // workgroup_mask=0, data_size=3 (8B)
  g1[1] = (int)((1024u & 0xFFFFu) << 16);       // tensor_dim0[15:0]=1024
  g1[2] = (int)(1u << 16);                      // tensor_dim0[31:16]=0 | tensor_dim1[15:0]=1
  g1[3] = (int)(1024u << 16);                   // tensor_dim1[31:16]=0 | tile_dim0=1024
  g1[4] = 0;                                    // tile_dim1=0 (1D), tile_dim2=0
  g1[5] = 1024;                                 // tensor_dim0_stride[31:0]
  g1[6] = 0;                                    // stride hi, tensor_dim1_stride lo
  g1[7] = 0;
  v4i g2 = {0, 0, 0, 0};
  v4i g3 = {0, 0, 0, 0};
  v8i g4 = {0, 0, 0, 0, 0, 0, 0, 0};
  __builtin_amdgcn_tensor_load_to_lds(g0, g1, g2, g3, g4, 0);
}

// ---- Kernel 0: zero accumulators ----
__global__ void ibq_init(float* __restrict__ avg_probs, float* __restrict__ scal) {
  int i = blockIdx.x * blockDim.x + threadIdx.x;
  if (i < NN) avg_probs[i] = 0.0f;
  if (i < 8) scal[i] = 0.0f;
}

// ---- Kernel 0b: codebook f32 -> bf16 ----
__global__ void ibq_cvt(const float* __restrict__ cb, __bf16* __restrict__ cbb) {
  int i = blockIdx.x * blockDim.x + threadIdx.x;
  cbb[i] = (__bf16)cb[i];
}

// ---- Kernel 1: z = x @ Wp^T + bp  (WMMA bf16, one wave per 16x16 tile) ----
__global__ void ibq_proj(const float* __restrict__ x, const float* __restrict__ Wp,
                         const float* __restrict__ bp, float* __restrict__ zf,
                         __bf16* __restrict__ zb) {
  int wave = (blockIdx.x * blockDim.x + threadIdx.x) >> 5;   // 0..4095
  int lane = threadIdx.x & 31;
  int tm = wave >> 4;            // M-tile 0..255
  int tn = wave & 15;            // N-tile 0..15
  int h = lane >> 4, ln = lane & 15;
  const float* xrow = x  + (size_t)(tm * 16 + ln) * DIMK;
  const float* wrow = Wp + (size_t)(tn * 16 + ln) * DIMK;
  v8f c = {};
  for (int k0 = 0; k0 < DIMK; k0 += 32) {
    v16bf a = load_frag_f32(xrow, k0, h);
    v16bf b = load_frag_f32(wrow, k0, h);
    c = wmma_bf16(a, b, c);
  }
  float bias = bp[tn * 16 + ln];
#pragma unroll
  for (int r = 0; r < 8; r++) {
    int m = tm * 16 + r + 8 * h;       // D layout: VGPR r -> M=r (+8 for upper half)
    int n = tn * 16 + ln;
    float v = c[r] + bias;
    zf[(size_t)m * ZD + n] = v;
    zb[(size_t)m * ZD + n] = (__bf16)v;
  }
}

// ---- Kernel 2: pass 1 — online softmax stats with TDM double-buffered B tiles ----
__global__ void __launch_bounds__(256)
ibq_pass1(const __bf16* __restrict__ zb, const __bf16* __restrict__ cb,
          float* __restrict__ row_max, float* __restrict__ row_sum,
          int* __restrict__ row_arg) {
  __shared__ __align__(16) __bf16 lds_b[8][2][16 * ZD];   // 128 KB: per-wave double buffers
  __shared__ float lm[8][16], ls[8][16];
  __shared__ int   lg[8][16];

  int mt = blockIdx.x;                 // one 16-row tile per block
  int wid = threadIdx.x >> 5;          // 8 waves
  int lane = threadIdx.x & 31;
  int h = lane >> 4, ln = lane & 15;
  const __bf16* zrow = zb + (size_t)(mt * 16 + ln) * ZD;
  v16bf a[8];
#pragma unroll
  for (int i = 0; i < 8; i++) a[i] = load_frag_bf16(zrow, i * 32, h);   // K=256 resident

  unsigned int lds_addr[2];
  lds_addr[0] = (unsigned int)(uintptr_t)&lds_b[wid][0][0];   // flat LDS aperture: low 32 bits = byte offset
  lds_addr[1] = (unsigned int)(uintptr_t)&lds_b[wid][1][0];

  float m_[8], s_[8]; int g_[8];
#pragma unroll
  for (int r = 0; r < 8; r++) { m_[r] = -3.0e38f; s_[r] = 0.0f; g_[r] = 0; }

  // prologue: fetch first tile for this wave
  tdm_load_tile(cb + (size_t)wid * 16 * ZD, lds_addr[0]);

  int buf = 0;
  for (int nt = wid; nt < NT; nt += 8, buf ^= 1) {
    if (nt + 8 < NT) {
      tdm_load_tile(cb + (size_t)(nt + 8) * 16 * ZD, lds_addr[buf ^ 1]);
      __builtin_amdgcn_s_wait_tensorcnt(1);      // TDM in-order per wave: current buf ready
    } else {
      __builtin_amdgcn_s_wait_tensorcnt(0);
    }
    const __bf16* crow = &lds_b[wid][buf][ln * ZD];
    v8f c = {};
#pragma unroll
    for (int i = 0; i < 8; i++) {
      v16bf b = load_frag_bf16(crow, i * 32, h);
      c = wmma_bf16(a[i], b, c);
    }
    int n = nt * 16 + ln;
#pragma unroll
    for (int r = 0; r < 8; r++) {
      float v = c[r] * TEMP_INV;                       // lt = logits / TEMP
      if (v > m_[r]) { s_[r] = s_[r] * __expf(m_[r] - v) + 1.0f; m_[r] = v; g_[r] = n; }
      else            s_[r] += __expf(v - m_[r]);
    }
  }
  // reduce (m,s,arg) across the 16 lanes of each half-wave
#pragma unroll
  for (int r = 0; r < 8; r++) {
#pragma unroll
    for (int mask = 1; mask < 16; mask <<= 1) {
      float om = __shfl_xor(m_[r], mask, 32);
      float os = __shfl_xor(s_[r], mask, 32);
      int   og = __shfl_xor(g_[r], mask, 32);
      float Mn = fmaxf(m_[r], om);
      s_[r] = s_[r] * __expf(m_[r] - Mn) + os * __expf(om - Mn);
      if (om > m_[r]) g_[r] = og;
      m_[r] = Mn;
    }
  }
  if (ln == 0) {                                  // lanes 0 (rows 0-7) and 16 (rows 8-15)
#pragma unroll
    for (int r = 0; r < 8; r++) { lm[wid][h * 8 + r] = m_[r]; ls[wid][h * 8 + r] = s_[r]; lg[wid][h * 8 + r] = g_[r]; }
  }
  __syncthreads();
  if (threadIdx.x < 16) {
    int row = threadIdx.x;
    float M = lm[0][row], S = ls[0][row]; int G = lg[0][row];
    for (int w = 1; w < 8; w++) {
      float om = lm[w][row], os = ls[w][row]; int og = lg[w][row];
      float Mn = fmaxf(M, om);
      S = S * __expf(M - Mn) + os * __expf(om - Mn);
      if (om > M) G = og;
      M = Mn;
    }
    row_max[mt * 16 + row] = M;
    row_sum[mt * 16 + row] = S;
    row_arg[mt * 16 + row] = G;
  }
}

// ---- Kernel 3: pass 2 — probs, avg_probs (64KB LDS histogram), sample-entropy acc ----
__global__ void __launch_bounds__(256)
ibq_pass2(const __bf16* __restrict__ zb, const __bf16* __restrict__ cb,
          const float* __restrict__ row_max, const float* __restrict__ row_sum,
          float* __restrict__ avg_probs, float* __restrict__ se_acc) {
  __shared__ float lds_avg[NN];                     // 64 KB of the 320 KB WGP LDS
  for (int i = threadIdx.x; i < NN; i += 256) lds_avg[i] = 0.0f;
  __syncthreads();

  int mt = blockIdx.x;
  int wid = threadIdx.x >> 5;
  int lane = threadIdx.x & 31;
  int h = lane >> 4, ln = lane & 15;
  const __bf16* zrow = zb + (size_t)(mt * 16 + ln) * ZD;
  v16bf a[8];
#pragma unroll
  for (int i = 0; i < 8; i++) a[i] = load_frag_bf16(zrow, i * 32, h);

  float m_[8], is_[8];
#pragma unroll
  for (int r = 0; r < 8; r++) {
    int row = mt * 16 + r + 8 * h;
    m_[r] = row_max[row];
    is_[r] = 1.0f / row_sum[row];
  }
  float spv = 0.0f;                                 // sum p * lt (this lane's share)

  for (int nt = wid; nt < NT; nt += 8) {
    const __bf16* crow = cb + (size_t)(nt * 16 + ln) * ZD;
    if (nt + 8 < NT) __builtin_prefetch(cb + (size_t)((nt + 8) * 16 + ln) * ZD, 0, 1);
    v8f c = {};
#pragma unroll
    for (int i = 0; i < 8; i++) {
      v16bf b = load_frag_bf16(crow, i * 32, h);
      c = wmma_bf16(a[i], b, c);
    }
    int n = nt * 16 + ln;
    float colp = 0.0f;
#pragma unroll
    for (int r = 0; r < 8; r++) {
      float v = c[r] * TEMP_INV;
      float p = __expf(v - m_[r]) * is_[r];
      spv += p * v;
      colp += p;
    }
    colp += __shfl_xor(colp, 16, 32);               // combine row halves (same n)
    if (h == 0) lds_avg[n] += colp;                 // waves own disjoint n -> no atomics
  }
  // sample-entropy: accumulate -sum p*lt; (m + log s) added in final kernel
#pragma unroll
  for (int mask = 1; mask < 32; mask <<= 1) spv += __shfl_xor(spv, mask, 32);
  if (lane == 0) atomicAdd(se_acc, -spv);

  __syncthreads();
  for (int i = threadIdx.x; i < NN; i += 256) atomicAdd(&avg_probs[i], lds_avg[i]);
}

// ---- Kernel 4: quant = codebook[idx], indices out, MSE accumulation ----
__global__ void __launch_bounds__(256)
ibq_quant(const float* __restrict__ codebook, const float* __restrict__ zf,
          const int* __restrict__ row_arg, float* __restrict__ out_quant,
          int* __restrict__ out_idx, float* __restrict__ mse_acc) {
  int m = blockIdx.x, t = threadIdx.x;
  int idx = row_arg[m];
  if (t == 0) out_idx[m] = idx;
  float q = codebook[(size_t)idx * ZD + t];
  out_quant[(size_t)m * ZD + t] = q;
  float d = q - zf[(size_t)m * ZD + t];
  float e = d * d;
#pragma unroll
  for (int mask = 1; mask < 32; mask <<= 1) e += __shfl_xor(e, mask, 32);
  __shared__ float red[8];
  if ((t & 31) == 0) red[t >> 5] = e;
  __syncthreads();
  if (t == 0) {
    float s = 0.0f;
    for (int i = 0; i < 8; i++) s += red[i];
    atomicAdd(mse_acc, s);
  }
}

// ---- Kernel 5: final loss ----
__global__ void __launch_bounds__(256)
ibq_final(const float* __restrict__ avg_probs, const float* __restrict__ row_max,
          const float* __restrict__ row_sum, const float* __restrict__ scal,
          float* __restrict__ out_loss) {
  __shared__ float r1[256], r2[256];
  float ae = 0.0f;
  for (int i = threadIdx.x; i < NN; i += 256) {
    float p = avg_probs[i] * (1.0f / (float)MM);
    ae += -p * __logf(p + EPSF);
  }
  float mls = 0.0f;
  for (int i = threadIdx.x; i < MM; i += 256)
    mls += row_max[i] + __logf(row_sum[i]);
  r1[threadIdx.x] = ae;
  r2[threadIdx.x] = mls;
  __syncthreads();
  for (int s = 128; s > 0; s >>= 1) {
    if (threadIdx.x < s) { r1[threadIdx.x] += r1[threadIdx.x + s]; r2[threadIdx.x] += r2[threadIdx.x + s]; }
    __syncthreads();
  }
  if (threadIdx.x == 0) {
    float mse = scal[0] / ((float)MM * (float)ZD);
    float sample_entropy = (r2[0] + scal[1]) / (float)MM;   // mean over rows of (m+log s - sum p*lt)
    float loss = 2.25f * mse + 0.05f * (sample_entropy - r1[0]);
    out_loss[0] = loss;
  }
}

extern "C" void kernel_launch(void* const* d_in, const int* in_sizes, int n_in,
                              void* d_out, int out_size, void* d_ws, size_t ws_size,
                              hipStream_t stream) {
  (void)in_sizes; (void)n_in; (void)out_size; (void)ws_size;
  const float* x  = (const float*)d_in[0];
  const float* Wp = (const float*)d_in[1];
  const float* bp = (const float*)d_in[2];
  const float* cb = (const float*)d_in[3];

  char* ws = (char*)d_ws;
  float*  zf        = (float*)  (ws + 0);            // 4 MB   [4096,256] f32
  __bf16* zb        = (__bf16*) (ws + 4194304);      // 2 MB   [4096,256] bf16
  __bf16* cbb       = (__bf16*) (ws + 6291456);      // 8 MB   [16384,256] bf16
  float*  row_max   = (float*)  (ws + 14680064);     // 16 KB
  float*  row_sum   = (float*)  (ws + 14696448);     // 16 KB
  int*    row_arg   = (int*)    (ws + 14712832);     // 16 KB
  float*  avg_probs = (float*)  (ws + 14729216);     // 64 KB
  float*  scal      = (float*)  (ws + 14794752);     // [0]=mse acc, [1]=se acc

  float* out   = (float*)d_out;                      // quant [4096*256]
  int*   o_idx = (int*)(out + (size_t)MM * ZD);      // indices [4096]
  float* o_loss = out + (size_t)MM * ZD + MM;        // loss [1]

  ibq_init <<<64, 256, 0, stream>>>(avg_probs, scal);
  ibq_cvt  <<<(NN * ZD) / 256, 256, 0, stream>>>(cb, cbb);
  ibq_proj <<<1024, 128, 0, stream>>>(x, Wp, bp, zf, zb);       // 4096 waves, 1 tile each
  ibq_pass1<<<MM / 16, 256, 0, stream>>>(zb, cbb, row_max, row_sum, row_arg);
  ibq_pass2<<<MM / 16, 256, 0, stream>>>(zb, cbb, row_max, row_sum, avg_probs, &scal[1]);
  ibq_quant<<<MM, 256, 0, stream>>>(cb, zf, row_arg, out, o_idx, &scal[0]);
  ibq_final<<<1, 256, 0, stream>>>(avg_probs, row_max, row_sum, scal, o_loss);
}
